// NeRFDecoder_60705067761935
// MI455X (gfx1250) — matrix operand order repaired and checked by reading, not compiled, not run on hardware
//
#include <hip/hip_runtime.h>
#include <hip/hip_bf16.h>
#include <math.h>

typedef __attribute__((ext_vector_type(16))) _Float16     v16h;
typedef __attribute__((ext_vector_type(8)))  float        v8f;
typedef __attribute__((ext_vector_type(8)))  unsigned int v8u;

// ---------------------------------------------------------------------------
// Layout constants
// ---------------------------------------------------------------------------
#define LDSW 224   // per-row LDS width in halves: [x_pad 96 | h 128] = 224

// f16 weight pool offsets inside d_ws (units: halves)
#define OFF_P0 0        /* 96*128  */
#define OFF_P1 12288    /* 128*128 */
#define OFF_P2 28672
#define OFF_P3 45056
#define OFF_P4 61440
#define OFF_P5 77824    /* 224*128 */
#define OFF_P6 106496
#define OFF_P7 122880
#define OFF_FT 139264   /* 128*128 */
#define OFF_VW 155648   /* 192*64  */
// total = 167,936 halves = 335,872 bytes of d_ws

struct NerfParams {
  const float* rp;        // ray_points      [N,39]
  const float* sf;        // sampled_features[2,N,32]
  const float* rd;        // ray_directions  [N,27]
  const _Float16* w16;    // converted, padded f16 weights (d_ws)
  const float* pb[8];     // pts biases [128] x 8
  const float* fb;        // feat bias [128]
  const float* aw;        // alpha w [128]
  const float* ab;        // alpha b [1]
  const float* vb;        // views bias [64]
  const float* rw;        // rgb w [3,64]
  const float* rb;        // rgb b [3]
  float* out;             // [0,3N): rgb ; [3N,4N): sigma
  int N;
};

// ---------------------------------------------------------------------------
// Weight conversion: f32 row-major [fanOut][Ksrc] -> f16 padded [fanOut][Kp]
// ---------------------------------------------------------------------------
__global__ void nerf_conv_w(const float* __restrict__ src, _Float16* __restrict__ dst,
                            int fanOut, int Ksrc, int a_len, int a_pad, int b_len, int Kp) {
  int idx = blockIdx.x * blockDim.x + threadIdx.x;
  int total = fanOut * Kp;
  if (idx >= total) return;
  int n = idx / Kp, k = idx % Kp;
  float v = 0.f;
  if (k < a_len)                            v = src[n * Ksrc + k];
  else if (k >= a_pad && k < a_pad + b_len) v = src[n * Ksrc + a_len + (k - a_pad)];
  dst[idx] = (_Float16)v;
}

// ---------------------------------------------------------------------------
// One wave computes  D[RT*16 x NOUT] = relu?( A[RT*16 x KP] * W^T + bias )
// starting at LDS row `rbase`. RT 16-row A tiles -> each B fragment load
// feeds RT WMMAs (cuts L2 B traffic by RT). All A fragments are loaded to
// VGPRs BEFORE any store, so the output LDS region may overlap the input
// region (per-wave DS ops are in-order).
// ---------------------------------------------------------------------------
template <int KP, int NOUT, bool RELU, int RT>
__device__ __forceinline__ void wave_gemm(_Float16* __restrict__ L, int rbase, int in_c0,
                                          const _Float16* __restrict__ wt,
                                          const float* __restrict__ bias,
                                          int out_c0, int lane) {
  constexpr int NFRAG = KP / 32;
  const int m = lane & 15;   // A: row (within tile) / B,D: col
  const int g = lane >> 4;   // lane half-group

  // --- A fragments: contiguous k = 16g + s order (same as B) -------------
  v16h a[RT][NFRAG];
#pragma unroll
  for (int rt = 0; rt < RT; ++rt) {
#pragma unroll
    for (int t = 0; t < NFRAG; ++t) {
      v8u tmp;
#pragma unroll
      for (int pp = 0; pp < 8; ++pp) {
        const int base = 2 * pp + 16 * g; // pair -> K=base,base+1
        tmp[pp] = *(const unsigned int*)(L + (rbase + rt * 16 + m) * LDSW +
                                         in_c0 + t * 32 + base);
      }
      a[rt][t] = __builtin_bit_cast(v16h, tmp);
    }
  }

  // --- per 16-col output tile: accumulate over K, add bias, relu, store ---
#pragma unroll
  for (int n0 = 0; n0 < NOUT; n0 += 16) {
    v8f acc[RT] = {};
    const _Float16* wp = wt + (size_t)(n0 + m) * KP + g * 16;  // B: N=m, K=k0+16g..
#pragma unroll
    for (int t = 0; t < NFRAG; ++t) {
      v16h b = *(const v16h*)(wp + t * 32);
#pragma unroll
      for (int rt = 0; rt < RT; ++rt)
        acc[rt] = __builtin_amdgcn_wmma_f32_16x16x32_f16(false, a[rt][t], false, b,
                                                         (short)0, acc[rt], false, false);
    }
    const float bn = bias[n0 + m];
#pragma unroll
    for (int rt = 0; rt < RT; ++rt) {
#pragma unroll
      for (int r = 0; r < 8; ++r) {          // D: row = rt*16 + r + 8g
        float v = acc[rt][r] + bn;
        if (RELU) v = v > 0.f ? v : 0.f;
        L[(rbase + rt * 16 + r + 8 * g) * LDSW + out_c0 + n0 + m] = (_Float16)v;
      }
    }
  }
}

// ---------------------------------------------------------------------------
// Fused NeRF decoder: 64 threads = 2 waves, 64 points per wave.
// ---------------------------------------------------------------------------
__global__ __launch_bounds__(64) void nerf_fused(NerfParams p) {
  __shared__ _Float16 smem[2][64][LDSW];   // 57,344 B
  const int tid = threadIdx.x;
  const int wave = tid >> 5, lane = tid & 31;
  _Float16* L = &smem[wave][0][0];
  const long row0 = (long)blockIdx.x * 128 + (long)wave * 64;

  // ---- stage x_pad = [ray_points(39) | sf0(32) | zeros(25)] in cols 0..95
  for (int i = lane; i < 64 * 96; i += 32) {
    const int r = i / 96, c = i % 96;
    const long gr = row0 + r;
    float v = 0.f;
    if (c < 39)      v = p.rp[gr * 39 + c];
    else if (c < 71) v = p.sf[gr * 32 + (c - 39)];
    L[r * LDSW + c] = (_Float16)v;
  }
  asm volatile("" ::: "memory");

  // ---- pts MLP (h lives in cols 96..223) ---------------------------------
  wave_gemm< 96, 128, true, 4>(L, 0,  0, p.w16 + OFF_P0, p.pb[0], 96, lane);
  asm volatile("" ::: "memory");
  wave_gemm<128, 128, true, 4>(L, 0, 96, p.w16 + OFF_P1, p.pb[1], 96, lane);
  asm volatile("" ::: "memory");
  wave_gemm<128, 128, true, 4>(L, 0, 96, p.w16 + OFF_P2, p.pb[2], 96, lane);
  asm volatile("" ::: "memory");
  wave_gemm<128, 128, true, 4>(L, 0, 96, p.w16 + OFF_P3, p.pb[3], 96, lane);
  asm volatile("" ::: "memory");
  wave_gemm<128, 128, true, 4>(L, 0, 96, p.w16 + OFF_P4, p.pb[4], 96, lane);
  asm volatile("" ::: "memory");
  // skip layer: input = [x_pad(96) | h(128)] = cols 0..223, K padded to 224.
  // RT=2 x2 (row halves) keeps A-register pressure < 256 VGPRs.
  wave_gemm<224, 128, true, 2>(L,  0, 0, p.w16 + OFF_P5, p.pb[5], 96, lane);
  asm volatile("" ::: "memory");
  wave_gemm<224, 128, true, 2>(L, 32, 0, p.w16 + OFF_P5, p.pb[5], 96, lane);
  asm volatile("" ::: "memory");
  wave_gemm<128, 128, true, 4>(L, 0, 96, p.w16 + OFF_P6, p.pb[6], 96, lane);
  asm volatile("" ::: "memory");
  wave_gemm<128, 128, true, 4>(L, 0, 96, p.w16 + OFF_P7, p.pb[7], 96, lane);
  asm volatile("" ::: "memory");

  // ---- sigma = h . alpha_w + alpha_b (2 rows per lane) -------------------
#pragma unroll
  for (int rr = 0; rr < 2; ++rr) {
    const int m = lane + rr * 32;
    float s = p.ab[0];
    for (int k = 0; k < 128; ++k)
      s += (float)L[m * LDSW + 96 + k] * p.aw[k];
    p.out[(long)3 * p.N + row0 + m] = s;
  }
  asm volatile("" ::: "memory");

  // ---- feature = h @ feat_w^T + feat_b (no relu) -> cols 0..127 ----------
  wave_gemm<128, 128, false, 4>(L, 0, 96, p.w16 + OFF_FT, p.fb, 0, lane);
  asm volatile("" ::: "memory");

  // ---- cols 128..191 = [dirs(27) | zeros(5) | sf1(32)] -------------------
  for (int i = lane; i < 64 * 64; i += 32) {
    const int r = i / 64, c = i % 64;
    const long gr = row0 + r;
    float v = 0.f;
    if (c < 27)       v = p.rd[gr * 27 + c];
    else if (c >= 32) v = p.sf[((long)p.N + gr) * 32 + (c - 32)];
    L[r * LDSW + 128 + c] = (_Float16)v;
  }
  asm volatile("" ::: "memory");

  // ---- h2 = relu([feature|dirs|sf1] @ views_w^T + vb) -> cols 0..63 ------
  wave_gemm<192, 64, true, 2>(L,  0, 0, p.w16 + OFF_VW, p.vb, 0, lane);
  asm volatile("" ::: "memory");
  wave_gemm<192, 64, true, 2>(L, 32, 0, p.w16 + OFF_VW, p.vb, 0, lane);
  asm volatile("" ::: "memory");

  // ---- rgb = affine(sigmoid(h2 @ rgb_w^T + rb)) (2 rows per lane) --------
#pragma unroll
  for (int rr = 0; rr < 2; ++rr) {
    const int m = lane + rr * 32;
    const long gr = row0 + m;
    for (int c = 0; c < 3; ++c) {
      float s = p.rb[c];
      for (int k = 0; k < 64; ++k)
        s += (float)L[m * LDSW + k] * p.rw[c * 64 + k];
      const float sg = 1.0f / (1.0f + __expf(-s));
      p.out[gr * 3 + c] = sg * 1.002f - 0.001f;
    }
  }
}

// ---------------------------------------------------------------------------
extern "C" void kernel_launch(void* const* d_in, const int* in_sizes, int n_in,
                              void* d_out, int out_size, void* d_ws, size_t ws_size,
                              hipStream_t stream) {
  (void)n_in; (void)out_size; (void)ws_size;

  const float* rp = (const float*)d_in[0];
  const float* sf = (const float*)d_in[1];
  const float* rd = (const float*)d_in[2];
  const float* pw[8]; const float* pb[8];
  for (int i = 0; i < 8; ++i) pw[i] = (const float*)d_in[3 + i];
  for (int i = 0; i < 8; ++i) pb[i] = (const float*)d_in[11 + i];
  const float* vw = (const float*)d_in[19];
  const float* vb = (const float*)d_in[20];
  const float* fw = (const float*)d_in[21];
  const float* fb = (const float*)d_in[22];
  const float* aw = (const float*)d_in[23];
  const float* ab = (const float*)d_in[24];
  const float* rw = (const float*)d_in[25];
  const float* rb = (const float*)d_in[26];

  const int N = in_sizes[0] / 39;
  _Float16* w16 = (_Float16*)d_ws;

  // ---- convert + zero-pad weights to f16 pool ----------------------------
  struct WCfg { const float* src; int off, fo, ks, al, ap, bl, kp; };
  const WCfg cfgs[10] = {
    { pw[0], OFF_P0, 128,  71,  71,  96,   0,  96 },
    { pw[1], OFF_P1, 128, 128, 128, 128,   0, 128 },
    { pw[2], OFF_P2, 128, 128, 128, 128,   0, 128 },
    { pw[3], OFF_P3, 128, 128, 128, 128,   0, 128 },
    { pw[4], OFF_P4, 128, 128, 128, 128,   0, 128 },
    { pw[5], OFF_P5, 128, 199,  71,  96, 128, 224 },  // [x(71)|pad|h(128)]
    { pw[6], OFF_P6, 128, 128, 128, 128,   0, 128 },
    { pw[7], OFF_P7, 128, 128, 128, 128,   0, 128 },
    { fw,    OFF_FT, 128, 128, 128, 128,   0, 128 },
    { vw,    OFF_VW,  64, 187, 155, 160,  32, 192 },  // [feat128|dirs27|pad|sf32]
  };
  for (int i = 0; i < 10; ++i) {
    const int total = cfgs[i].fo * cfgs[i].kp;
    nerf_conv_w<<<(total + 255) / 256, 256, 0, stream>>>(
        cfgs[i].src, w16 + cfgs[i].off, cfgs[i].fo, cfgs[i].ks,
        cfgs[i].al, cfgs[i].ap, cfgs[i].bl, cfgs[i].kp);
  }

  // ---- fused MLP ---------------------------------------------------------
  NerfParams P;
  P.rp = rp; P.sf = sf; P.rd = rd; P.w16 = (const _Float16*)w16;
  for (int i = 0; i < 8; ++i) P.pb[i] = pb[i];
  P.fb = fb; P.aw = aw; P.ab = ab; P.vb = vb; P.rw = rw; P.rb = rb;
  P.out = (float*)d_out; P.N = N;

  nerf_fused<<<N / 128, 64, 0, stream>>>(P);
}
